// MultiHeadAttention_25512105738552
// MI455X (gfx1250) — compile-verified, matrix operand order checked by reading, not run on hardware
//
#include <hip/hip_runtime.h>
#include <cstdint>

typedef __attribute__((ext_vector_type(16))) __bf16 v16bf;
typedef __attribute__((ext_vector_type(8)))  __bf16 v8bf;
typedef __attribute__((ext_vector_type(8)))  float  v8f;
typedef __attribute__((ext_vector_type(4)))  int    v4i;

#define DMODEL 1024
#define SEQ    1024
#define NB     8
#define NHD    16
#define DK     64

#if defined(__gfx1250__) && __has_builtin(__builtin_amdgcn_global_load_async_to_lds_b128)
#define USE_ASYNC_LDS 1
// global (AS1) / LDS (AS3) vector-typed pointers for the async builtin
#define GV4I(p) ((__attribute__((address_space(1))) v4i*)(p))
#define LV4I(p) ((__attribute__((address_space(3))) v4i*)(p))
#if __has_builtin(__builtin_amdgcn_s_wait_asynccnt)
#define WAIT_ASYNC(n) __builtin_amdgcn_s_wait_asynccnt(n)
#else
#define WAIT_ASYNC(n) asm volatile("s_wait_asynccnt %0" ::"i"(n) : "memory")
#endif
#else
#define USE_ASYNC_LDS 0
#endif

static __device__ inline v8f wmma_bf16f32(v16bf a, v16bf b, v8f c) {
  return __builtin_amdgcn_wmma_f32_16x16x32_bf16(false, a, false, b, (short)0, c,
                                                 false, false);
}

// Element e of a 16-element bf16 A/B fragment holds K-index:
// lanes 0-15: K = {kb..kb+7, 16+kb..16+kb+7} with kb=0; lanes 16-31: kb=8.
// Both runs are contiguous -> two 16-byte loads per fragment.
static __device__ inline int frag_k(int e, int kb) {
  return (e < 8) ? (kb + e) : (16 + kb + (e - 8));
}

// A fragment from row-major [16 x >=32] (row = lane&15, row stride ld elements).
// Also a B fragment when the source is [N][K] (n-major, K contiguous).
// Requires base 16B-aligned and ld*2 a multiple of 16 bytes.
static __device__ inline v16bf load_frag_k16(const __bf16* base, int ld) {
  int lane = threadIdx.x & 31;
  int rc = lane & 15;
  int kb = (lane >> 4) << 3;
  const __bf16* p = base + rc * ld + kb;
  v8bf lo = *(const v8bf*)p;
  v8bf hi = *(const v8bf*)(p + 16);
  return __builtin_shufflevector(lo, hi, 0, 1, 2, 3, 4, 5, 6, 7,
                                 8, 9, 10, 11, 12, 13, 14, 15);
}

// B fragment from k-major storage [32(k) x ncols] (col = lane&15, row stride ld).
static __device__ inline v16bf load_frag_kmajor(const __bf16* base, int ld) {
  int lane = threadIdx.x & 31;
  int col = lane & 15;
  int kb = (lane >> 4) << 3;
  v16bf f;
#pragma unroll
  for (int e = 0; e < 16; ++e)
    f[e] = base[frag_k(e, kb) * ld + col];
  return f;
}

__global__ __launch_bounds__(256) void cvt_f32_bf16(const float* __restrict__ in,
                                                    __bf16* __restrict__ out, int n) {
  int i = (blockIdx.x * 256 + threadIdx.x) * 8;
  if (i + 8 <= n) {
    v8f f = *(const v8f*)(in + i);
    v8bf o;
#pragma unroll
    for (int e = 0; e < 8; ++e) o[e] = (__bf16)f[e];
    *(v8bf*)(out + i) = o;
  }
}

// Stage one 128x32 A tile + 64x32 B tile into LDS buffers.
// Async path: 3 x global_load_async_to_lds_b128 per thread (ASYNCcnt += 3).
static __device__ inline void stage_tile(const __bf16* __restrict__ A,
                                         const __bf16* __restrict__ W,
                                         __bf16 (*As)[32], __bf16 (*Bs)[32],
                                         int m0, int n0, int K, int k0, int tid) {
#if USE_ASYNC_LDS
  int r = tid >> 2, c = (tid & 3) << 3;
  __builtin_amdgcn_global_load_async_to_lds_b128(
      GV4I(&A[(size_t)(m0 + r) * K + k0 + c]), LV4I(&As[r][c]), 0, 0);
  int r2 = (tid + 256) >> 2, c2 = ((tid + 256) & 3) << 3;
  __builtin_amdgcn_global_load_async_to_lds_b128(
      GV4I(&A[(size_t)(m0 + r2) * K + k0 + c2]), LV4I(&As[r2][c2]), 0, 0);
  __builtin_amdgcn_global_load_async_to_lds_b128(
      GV4I(&W[(size_t)(n0 + r) * K + k0 + c]), LV4I(&Bs[r][c]), 0, 0);
#else
  for (int i = tid; i < 128 * 4; i += 256) {
    int r = i >> 2, c = (i & 3) << 3;
    *(v8bf*)&As[r][c] = *(const v8bf*)&A[(size_t)(m0 + r) * K + k0 + c];
  }
  for (int i = tid; i < 64 * 4; i += 256) {
    int r = i >> 2, c = (i & 3) << 3;
    *(v8bf*)&Bs[r][c] = *(const v8bf*)&W[(size_t)(n0 + r) * K + k0 + c];
  }
#endif
}

// C = A[M,K] @ W[N,K]^T + bias.  Workgroup tile 128x64, 8 waves (2x4),
// each wave: 4 m-tiles x 1 n-tile of 16x16, bf16 WMMA, f32 accumulate.
// Double-buffered async LDS staging: loads for tile t+1 overlap WMMAs of
// tile t; s_wait_asynccnt(3) drains exactly tile t's loads (in-order per wave).
// mode 0: write bf16 to outB in [B,H,S,dk] layout (for Q/K/V)
// mode 1: write f32 row-major [M,N] to outF (final projection -> d_out)
__global__ __launch_bounds__(256) void gemm_bf16_wmma(
    const __bf16* __restrict__ A, const __bf16* __restrict__ W,
    const float* __restrict__ bias, __bf16* __restrict__ outB,
    float* __restrict__ outF, int M, int N, int K, int mode) {
  __shared__ __attribute__((aligned(16))) __bf16 As[2][128][32];
  __shared__ __attribute__((aligned(16))) __bf16 Bs[2][64][32];
  int tid = threadIdx.x;
  int wave = tid >> 5, lane = tid & 31;
  int wr = wave >> 2, wc = wave & 3;
  int m0 = blockIdx.y * 128;
  int n0 = blockIdx.x * 64;
  v8f acc[4] = {};

  stage_tile(A, W, As[0], Bs[0], m0, n0, K, 0, tid);

  for (int k0 = 0; k0 < K; k0 += 32) {
    int cur = (k0 >> 5) & 1;
    if (k0 + 32 < K) {
      stage_tile(A, W, As[cur ^ 1], Bs[cur ^ 1], m0, n0, K, k0 + 32, tid);
#if USE_ASYNC_LDS
      WAIT_ASYNC(3);   // tile t's 3 loads done; tile t+1's 3 stay in flight
#endif
    } else {
#if USE_ASYNC_LDS
      WAIT_ASYNC(0);
#endif
    }
    __syncthreads();   // all waves' staging of `cur` visible

    v16bf bf = load_frag_k16(&Bs[cur][wc * 16][0], 32);
#pragma unroll
    for (int t = 0; t < 4; ++t) {
      v16bf af = load_frag_k16(&As[cur][wr * 64 + t * 16][0], 32);
      acc[t] = wmma_bf16f32(af, bf, acc[t]);
    }
    __syncthreads();   // reads of `cur` done before it is restaged (t+2)
  }

  // C/D layout: n = lane&15, m = vgpr + 8*(lane>>4)
  int n = lane & 15, mo = (lane >> 4) << 3;
  int gn = n0 + wc * 16 + n;
  float bv = bias[gn];
#pragma unroll
  for (int t = 0; t < 4; ++t) {
#pragma unroll
    for (int r = 0; r < 8; ++r) {
      int gm = m0 + wr * 64 + t * 16 + mo + r;
      float v = acc[t][r] + bv;
      if (mode == 0) {
        int b = gm >> 10, s = gm & (SEQ - 1);
        int h = gn >> 6, d = gn & (DK - 1);
        outB[(((size_t)b * NHD + h) * SEQ + s) * DK + d] = (__bf16)v;
      } else {
        outF[(size_t)gm * N + gn] = v;
      }
    }
  }
}

// Fused scores + batch-axis softmax + PV.
// grid = (SEQ/16 q-tiles, NHD heads); block = 256 = 8 waves; wave b <-> batch b.
// Softmax is over the batch axis (8 values), pointwise in (q,k): exact per
// k-tile, so scores never leave LDS and PV fuses with no online rescaling.
__global__ __launch_bounds__(256) void attn_wmma(
    const __bf16* __restrict__ Q, const __bf16* __restrict__ K,
    const __bf16* __restrict__ V, __bf16* __restrict__ heads) {
  __shared__ __attribute__((aligned(32))) float sc[NB][16][32];  // raw scores
  __shared__ __attribute__((aligned(32))) float mxs[16][32];     // max over batch
  __shared__ __attribute__((aligned(32))) float rcs[16][32];     // 1/sum(exp)
  int h = blockIdx.y;
  int q0 = blockIdx.x * 16;
  int tid = threadIdx.x;
  int b = tid >> 5;
  int lane = tid & 31;

  const __bf16* Qb = Q + (((size_t)b * NHD + h) * SEQ + q0) * DK;
  const __bf16* Kb = K + ((size_t)b * NHD + h) * SEQ * DK;
  const __bf16* Vb = V + ((size_t)b * NHD + h) * SEQ * DK;

  // Q A-fragments for d-chunks [0,32) and [32,64): loop-invariant
  v16bf qa0 = load_frag_k16(Qb, DK);
  v16bf qa1 = load_frag_k16(Qb + 32, DK);

  v8f o[4] = {};   // 16(q) x 64(dk) output accumulator

  for (int k0 = 0; k0 < SEQ; k0 += 32) {
    // --- scores: two 16-wide k-tiles, each = Q(16x64) @ K-tile(64x16)^T
#pragma unroll
    for (int kt = 0; kt < 2; ++kt) {
      v8f s = {};
      v16bf kb0 = load_frag_k16(Kb + (size_t)(k0 + kt * 16) * DK, DK);
      s = wmma_bf16f32(qa0, kb0, s);
      v16bf kb1 = load_frag_k16(Kb + (size_t)(k0 + kt * 16) * DK + 32, DK);
      s = wmma_bf16f32(qa1, kb1, s);
      int n = lane & 15, mo = (lane >> 4) << 3;
#pragma unroll
      for (int r = 0; r < 8; ++r)
        sc[b][mo + r][kt * 16 + n] = s[r];
    }
    __syncthreads();

    // --- cooperative batch max / reciprocal-sum per (q,k) position
    for (int pos = tid; pos < 16 * 32; pos += 256) {
      int rr = pos >> 5, kk = pos & 31;
      float vals[NB];
      float m = -3.0e38f;
#pragma unroll
      for (int bb = 0; bb < NB; ++bb) {
        vals[bb] = sc[bb][rr][kk];
        m = fmaxf(m, vals[bb]);
      }
      float ssum = 0.f;
#pragma unroll
      for (int bb = 0; bb < NB; ++bb) ssum += __expf(vals[bb] - m);
      mxs[rr][kk] = m;
      rcs[rr][kk] = __builtin_amdgcn_rcpf(ssum);
    }
    __syncthreads();

    // --- P A-fragment (16x32 bf16): one exp + one mul per element,
    //     vectorized 32B LDS reads along each contiguous K-run
    int row = lane & 15;
    int kb = (lane >> 4) << 3;
    v8f s0 = *(const v8f*)&sc[b][row][kb];
    v8f s1 = *(const v8f*)&sc[b][row][16 + kb];
    v8f m0v = *(const v8f*)&mxs[row][kb];
    v8f m1v = *(const v8f*)&mxs[row][16 + kb];
    v8f r0v = *(const v8f*)&rcs[row][kb];
    v8f r1v = *(const v8f*)&rcs[row][16 + kb];
    v16bf p;
#pragma unroll
    for (int e = 0; e < 8; ++e) {
      p[e] = (__bf16)(__expf(s0[e] - m0v[e]) * r0v[e]);
      p[8 + e] = (__bf16)(__expf(s1[e] - m1v[e]) * r1v[e]);
    }

    // --- PV: o(16x64) += P(16x32) @ V(32x64), 4 dk-chunks of 16 cols
#pragma unroll
    for (int c = 0; c < 4; ++c) {
      v16bf vb = load_frag_kmajor(Vb + (size_t)k0 * DK + c * 16, DK);
      o[c] = wmma_bf16f32(p, vb, o[c]);
    }
    __syncthreads();   // WAR: next iteration rewrites sc/mxs/rcs
  }

  // heads layout [B, S, H, dk] == row-major [B*S, DMODEL] for the final GEMM
  int n = lane & 15, mo = (lane >> 4) << 3;
#pragma unroll
  for (int c = 0; c < 4; ++c)
#pragma unroll
    for (int r = 0; r < 8; ++r)
      heads[(((size_t)b * SEQ + q0 + mo + r) * NHD + h) * DK + c * 16 + n] =
          (__bf16)o[c][r];
}

extern "C" void kernel_launch(void* const* d_in, const int* in_sizes, int n_in,
                              void* d_out, int out_size, void* d_ws, size_t ws_size,
                              hipStream_t stream) {
  const float* x  = (const float*)d_in[0];
  const float* Wq = (const float*)d_in[1];
  const float* bq = (const float*)d_in[2];
  const float* Wk = (const float*)d_in[3];
  const float* bk = (const float*)d_in[4];
  const float* Wv = (const float*)d_in[5];
  const float* bv = (const float*)d_in[6];
  const float* Wo = (const float*)d_in[7];
  const float* bo = (const float*)d_in[8];
  float* out = (float*)d_out;

  size_t off = 0;
  char* base = (char*)d_ws;
  auto alloc = [&](size_t bytes) -> __bf16* {
    __bf16* p = (__bf16*)(base + off);
    off += (bytes + 255) & ~(size_t)255;
    return p;
  };
  const size_t actBytes = (size_t)NB * SEQ * DMODEL * 2;  // 16 MB
  const size_t wBytes   = (size_t)DMODEL * DMODEL * 2;    // 2 MB
  __bf16* xb  = alloc(actBytes);
  __bf16* wqb = alloc(wBytes);
  __bf16* wkb = alloc(wBytes);
  __bf16* wvb = alloc(wBytes);
  __bf16* wob = alloc(wBytes);
  __bf16* qb  = alloc(actBytes);
  __bf16* kb  = alloc(actBytes);
  __bf16* vb_ = alloc(actBytes);
  __bf16* hd  = alloc(actBytes);

  int nx = NB * SEQ * DMODEL;
  int nw = DMODEL * DMODEL;
  cvt_f32_bf16<<<(nx / 8 + 255) / 256, 256, 0, stream>>>(x, xb, nx);
  cvt_f32_bf16<<<(nw / 8 + 255) / 256, 256, 0, stream>>>(Wq, wqb, nw);
  cvt_f32_bf16<<<(nw / 8 + 255) / 256, 256, 0, stream>>>(Wk, wkb, nw);
  cvt_f32_bf16<<<(nw / 8 + 255) / 256, 256, 0, stream>>>(Wv, wvb, nw);
  cvt_f32_bf16<<<(nw / 8 + 255) / 256, 256, 0, stream>>>(Wo, wob, nw);

  dim3 ggrid(DMODEL / 64, (NB * SEQ) / 128);  // (16, 64)
  gemm_bf16_wmma<<<ggrid, 256, 0, stream>>>(xb, wqb, bq, qb, nullptr,
                                            NB * SEQ, DMODEL, DMODEL, 0);
  gemm_bf16_wmma<<<ggrid, 256, 0, stream>>>(xb, wkb, bk, kb, nullptr,
                                            NB * SEQ, DMODEL, DMODEL, 0);
  gemm_bf16_wmma<<<ggrid, 256, 0, stream>>>(xb, wvb, bv, vb_, nullptr,
                                            NB * SEQ, DMODEL, DMODEL, 0);

  attn_wmma<<<dim3(SEQ / 16, NHD), 256, 0, stream>>>(qb, kb, vb_, hd);

  gemm_bf16_wmma<<<ggrid, 256, 0, stream>>>(hd, wob, bo, nullptr, out,
                                            NB * SEQ, DMODEL, DMODEL, 1);
}
